// Multihead_Attention_13761075217004
// MI455X (gfx1250) — matrix operand order repaired and checked by reading, not compiled, not run on hardware
//
#include <hip/hip_runtime.h>
#include <hip/hip_bf16.h>

typedef __attribute__((ext_vector_type(16))) __bf16 v16bf;
typedef __attribute__((ext_vector_type(8)))  float  v8f;
typedef unsigned short u16;
typedef unsigned int   u32;
typedef __attribute__((ext_vector_type(4)))  u32    u32x4;

union Frag16 { uint4 q[2]; v16bf v; };

__device__ __forceinline__ u16 f2bf(float f) {
    u32 u = __builtin_bit_cast(u32, f);
    u32 r = (u + 0x7FFFu + ((u >> 16) & 1u)) >> 16;   // round-to-nearest-even
    return (u16)r;
}
__device__ __forceinline__ u16 u2bf(u32 u) {
    return (u16)((u + 0x7FFFu + ((u >> 16) & 1u)) >> 16);
}

// ---------------------------------------------------------------------------
// Kernel 0: Wk (f32 [512][512]) -> bf16, same [n][k] layout. One float4 / thread.
__global__ __launch_bounds__(256) void mha_wk_to_bf16(const float* __restrict__ Wk,
                                                      u16* __restrict__ out) {
    int idx = blockIdx.x * 256 + threadIdx.x;      // float4 index, 0..65535
    float4 x = reinterpret_cast<const float4*>(Wk)[idx];
    uint2 p;
    p.x = (u32)f2bf(x.x) | ((u32)f2bf(x.y) << 16);
    p.y = (u32)f2bf(x.z) | ((u32)f2bf(x.w) << 16);
    reinterpret_cast<uint2*>(out)[idx] = p;
}

// ---------------------------------------------------------------------------
// Kernel 1: v[j] = relu(Q . Wq[j] + bq[j]) * Wout[j/64] * (1/sqrt(64))
__global__ __launch_bounds__(512) void mha_make_v(const float* __restrict__ Q,
                                                  const float* __restrict__ Wq,
                                                  const float* __restrict__ bq,
                                                  const float* __restrict__ Wout,
                                                  float* __restrict__ v) {
    __shared__ float qs[512];
    const int j = threadIdx.x;
    qs[j] = Q[j];
    __syncthreads();
    const float* w = Wq + (size_t)j * 512;
    float s = bq[j];
    #pragma unroll 4
    for (int c = 0; c < 512; ++c) s += qs[c] * w[c];
    v[j] = fmaxf(s, 0.f) * Wout[j >> 6] * 0.125f;   // 1/sqrt(64) = 0.125
}

// ---------------------------------------------------------------------------
// Kernel 2: main fused GEMM + epilogue.
//   64 rows of K per block, N=512, Kdim=512, bf16 WMMA with f32 accum.
//   8 waves = 4 column-quads x 2 row-halves. Waves w and w+4 stream the SAME
//   B fragments (concurrently -> WGP-cache hits), so L2 sees one Wk pass per
//   64-row block. Per-wave loop nest is fully unrolled & flat (no LICM bait).
__global__ __launch_bounds__(256) void mha_kproj_logits(const float* __restrict__ Kg,
                                                        const u16* __restrict__ WkBf,
                                                        const float* __restrict__ bk,
                                                        const float* __restrict__ v,
                                                        const float* __restrict__ bout,
                                                        float* __restrict__ logits) {
    constexpr int ROWS = 64, KDIM = 512;
    constexpr int STRIDE = KDIM * 2 + 16;          // 1040 B row pitch (bank-conflict pad)
    __shared__ __align__(16) unsigned char ldsK[ROWS * STRIDE];
    __shared__ float ldsLogit[ROWS];

    const int tid  = threadIdx.x;
    const int row0 = blockIdx.x * ROWS;

    if (tid < ROWS) ldsLogit[tid] = 0.f;

    // --- Stage K tile: f32 global (non-temporal) -> bf16 LDS ---
    const u32x4* Kg4 = reinterpret_cast<const u32x4*>(Kg) + (size_t)row0 * (KDIM / 4);
    #pragma unroll
    for (int i = 0; i < (ROWS * KDIM / 4) / 256; ++i) {
        int idx = i * 256 + tid;
        int r  = idx >> 7;                         // row in tile
        int kc = idx & 127;                        // float4 chunk in row
        u32x4 x = __builtin_nontemporal_load(&Kg4[r * 128 + kc]);
        uint2 p;
        p.x = (u32)u2bf(x.x) | ((u32)u2bf(x.y) << 16);
        p.y = (u32)u2bf(x.z) | ((u32)u2bf(x.w) << 16);
        *reinterpret_cast<uint2*>(&ldsK[r * STRIDE + kc * 8]) = p;
    }
    __syncthreads();

    const int wave = tid >> 5;
    const int lane = tid & 31;
    const int hi   = lane >> 4;                    // half-wave select
    const int ln   = lane & 15;                    // column / row-in-subtile lane
    const int wq   = wave & 3;                     // column-quad: ntiles 4*wq..4*wq+3
    const int mh   = wave >> 2;                    // row-half: rows mh*32..mh*32+31

    // A base for this lane: rows (mh*32 + ms*16 + ln); ISA 16-bit A layout chunks
    const unsigned char* aBase = &ldsK[(mh * 32 + ln) * STRIDE + hi * 16];

    #pragma unroll
    for (int ms = 0; ms < 2; ++ms) {               // two 16-row subtiles per wave
        float rowacc[8];
        #pragma unroll
        for (int r = 0; r < 8; ++r) rowacc[r] = 0.f;

        #pragma unroll
        for (int nti = 0; nti < 4; ++nti) {        // wave's 4 N-tiles
            const int ntile = wq * 4 + nti;
            const int col   = ntile * 16 + ln;
            v8f acc = {};
            // B: lane holds column `col`, K chunks 8*hi..+8 and 16+8*hi..+8
            const uint4* bBase =
                reinterpret_cast<const uint4*>(WkBf + (size_t)col * KDIM + hi * 8);
            #pragma unroll
            for (int k = 0; k < 16; ++k) {         // Kdim = 16 * 32
                Frag16 a, b;
                const uint4* ap =
                    reinterpret_cast<const uint4*>(aBase + ms * 16 * STRIDE + k * 64);
                a.q[0] = ap[0];                    // K = kbase + 8*hi   .. +8
                a.q[1] = ap[2];                    // K = kbase+16+8*hi .. +8
                b.q[0] = bBase[k * 4];             // K = kbase + 8*hi   .. +8
                b.q[1] = bBase[k * 4 + 2];         // K = kbase+16+8*hi .. +8
                acc = __builtin_amdgcn_wmma_f32_16x16x32_bf16(
                          false, a.v, false, b.v, (short)0, acc, false, false);
            }
            // Epilogue for this tile: relu(acc + bk[col]) * v[col]
            const float bkc = bk[col];
            const float vc  = v[col];
            #pragma unroll
            for (int r = 0; r < 8; ++r) {          // element r -> local row r + 8*hi
                float t = acc[r] + bkc;
                rowacc[r] += fmaxf(t, 0.f) * vc;
            }
        }
        // Reduce over the 16 column-lanes of each half-wave
        #pragma unroll
        for (int r = 0; r < 8; ++r) {
            rowacc[r] += __shfl_xor(rowacc[r], 1, 32);
            rowacc[r] += __shfl_xor(rowacc[r], 2, 32);
            rowacc[r] += __shfl_xor(rowacc[r], 4, 32);
            rowacc[r] += __shfl_xor(rowacc[r], 8, 32);
        }
        if (ln == 0) {
            #pragma unroll
            for (int r = 0; r < 8; ++r)
                atomicAdd(&ldsLogit[(mh * 2 + ms) * 16 + hi * 8 + r], rowacc[r]);
        }
    }
    __syncthreads();
    if (tid < ROWS) logits[row0 + tid] = ldsLogit[tid] + bout[0];
}

// ---------------------------------------------------------------------------
// Kernel 3: single-block max + sum(exp) over the logits.
__global__ __launch_bounds__(1024) void mha_softmax_stats(const float* __restrict__ logits,
                                                          int n, float* __restrict__ stats) {
    __shared__ float red[1024];
    const int tid = threadIdx.x;
    float m = -3.402823466e38f;
    for (int i = tid; i < n; i += 1024) m = fmaxf(m, logits[i]);
    red[tid] = m; __syncthreads();
    for (int s = 512; s > 0; s >>= 1) {
        if (tid < s) red[tid] = fmaxf(red[tid], red[tid + s]);
        __syncthreads();
    }
    const float mx = red[0];
    __syncthreads();
    float sum = 0.f;
    for (int i = tid; i < n; i += 1024) sum += __expf(logits[i] - mx);
    red[tid] = sum; __syncthreads();
    for (int s = 512; s > 0; s >>= 1) {
        if (tid < s) red[tid] += red[tid + s];
        __syncthreads();
    }
    if (tid == 0) { stats[0] = mx; stats[1] = red[0]; }
}

// ---------------------------------------------------------------------------
// Kernel 4: normalize.
__global__ __launch_bounds__(256) void mha_softmax_norm(const float* __restrict__ logits,
                                                        const float* __restrict__ stats,
                                                        float* __restrict__ out, int n) {
    int i = blockIdx.x * 256 + threadIdx.x;
    if (i < n) out[i] = __expf(logits[i] - stats[0]) / stats[1];
}

// ---------------------------------------------------------------------------
extern "C" void kernel_launch(void* const* d_in, const int* in_sizes, int n_in,
                              void* d_out, int out_size, void* d_ws, size_t ws_size,
                              hipStream_t stream) {
    const float* Q    = (const float*)d_in[0];
    const float* K    = (const float*)d_in[1];
    const float* Wq   = (const float*)d_in[2];
    const float* bq   = (const float*)d_in[3];
    const float* Wk   = (const float*)d_in[4];
    const float* bk   = (const float*)d_in[5];
    const float* Wout = (const float*)d_in[6];
    const float* bout = (const float*)d_in[7];
    float* out = (float*)d_out;

    const int Tk = in_sizes[1] / 512;              // 65536

    // workspace layout
    char*  wsb    = (char*)d_ws;
    u16*   wkbf   = (u16*)wsb;                     // 512*512*2 = 524288 B
    float* v      = (float*)(wsb + 524288);        // 512*4    = 2048 B
    float* logits = (float*)(wsb + 526336);        // Tk*4 B
    float* stats  = logits + Tk;                   // 2 floats

    mha_wk_to_bf16 <<<256, 256, 0, stream>>>(Wk, wkbf);
    mha_make_v     <<<1, 512, 0, stream>>>(Q, Wq, bq, Wout, v);
    mha_kproj_logits<<<Tk / 64, 256, 0, stream>>>(K, wkbf, bk, v, bout, logits);
    mha_softmax_stats<<<1, 1024, 0, stream>>>(logits, Tk, stats);
    mha_softmax_norm<<<(Tk + 255) / 256, 256, 0, stream>>>(logits, stats, out, Tk);
}
